// AWQW4A16Linear_33990371180876
// MI455X (gfx1250) — matrix operand order repaired and checked by reading, not compiled, not run on hardware
//
#include <hip/hip_runtime.h>
#include <hip/hip_bf16.h>
#include <stdint.h>

#define AS1 __attribute__((address_space(1)))
#define AS3 __attribute__((address_space(3)))

typedef __attribute__((ext_vector_type(16))) __bf16 v16bf;
typedef __attribute__((ext_vector_type(8)))  __bf16 v8bf;
typedef __attribute__((ext_vector_type(8)))  float  v8f;
typedef __attribute__((ext_vector_type(4)))  int    v4i;

#if defined(__has_builtin)
#if __has_builtin(__builtin_amdgcn_global_load_async_to_lds_b128) && \
    __has_builtin(__builtin_amdgcn_s_wait_asynccnt)
#define USE_ASYNC 1
#endif
#endif

static constexpr int KDIM   = 4096;
static constexpr int NDIM   = 12288;
static constexpr int MDIM   = 4096;
static constexpr int MTILE  = 128;
static constexpr int NTILE  = 128;
static constexpr int KTILE  = 64;          // == GROUP_SIZE
static constexpr int NK     = KDIM / KTILE;
static constexpr int LDST   = 72;          // LDS row stride in halfs (16B aligned)

__device__ __forceinline__ unsigned short f2bf(float f) {
    union { float f; uint32_t u; } v; v.f = f;
    uint32_t r = v.u + 0x7FFFu + ((v.u >> 16) & 1u);   // round-to-nearest-even
    return (unsigned short)(r >> 16);
}
__device__ __forceinline__ float bf2f(unsigned short h) {
    union { uint32_t u; float f; } v; v.u = ((uint32_t)h) << 16;
    return v.f;
}

__global__ __launch_bounds__(256) void awq_w4a16_wmma(
    const unsigned short* __restrict__ x,       // bf16 [M][K]
    const int*            __restrict__ qw,      // int32 [N/4][K/2]
    const unsigned short* __restrict__ scales,  // bf16 [K/64][N]
    const unsigned short* __restrict__ zeros,   // bf16 [K/64][N]
    const unsigned short* __restrict__ bias,    // bf16 [N]
    unsigned short*       __restrict__ out)     // bf16 [M][N]
{
    __shared__ unsigned short As[2][MTILE][LDST];  // 36864 B
    __shared__ unsigned short Ws[2][NTILE][LDST];  // 36864 B

    const int tid  = threadIdx.x;
    const int lane = tid & 31;
    const int wave = tid >> 5;
    const int wm   = wave >> 2;      // 0..1  (64-row slab)
    const int wn   = wave & 3;       // 0..3  (32-col slab)
    const int r16  = lane & 15;
    const int hh   = lane >> 4;

    const int n0 = blockIdx.x * NTILE;
    const int m0 = blockIdx.y * MTILE;

    // ---- dequant tasks: 2 per thread; each task = 2 int32 -> 4 rows x 4 k ----
    int o4r[2], pr[2];
    const int* qrow[2];
    const unsigned short* scp[2];
    const unsigned short* zrp[2];
#pragma unroll
    for (int t = 0; t < 2; ++t) {
        int g  = tid * 2 + t;        // 0..511
        o4r[t] = g >> 4;             // 0..31
        pr[t]  = g & 15;             // 0..15
        qrow[t] = qw + (size_t)(n0 / 4 + o4r[t]) * (KDIM / 2) + pr[t] * 2;
        scp[t]  = scales + n0 + o4r[t] * 4;
        zrp[t]  = zeros  + n0 + o4r[t] * 4;
    }

    // ---- A-tile copy: 4 x 16B chunks per thread ----
    int arow[4], ach[4];
#pragma unroll
    for (int l = 0; l < 4; ++l) {
        int cid = tid + l * 256;
        arow[l] = cid >> 3;          // 0..127
        ach[l]  = (cid & 7) * 8;     // half offset
    }

    auto issue_a = [&](int kt, int buf) {
        const int k0 = kt * KTILE;
#pragma unroll
        for (int l = 0; l < 4; ++l) {
            const unsigned short* g = x + (size_t)(m0 + arow[l]) * KDIM + k0 + ach[l];
            unsigned short* lp = &As[buf][arow[l]][ach[l]];
#ifdef USE_ASYNC
            __builtin_amdgcn_global_load_async_to_lds_b128(
                (AS1 v4i*)g, (AS3 v4i*)lp, 0, 0);
#else
            *(float4*)lp = *(const float4*)g;
#endif
        }
    };

    int2  qn[2];
    uint2 scn[2], zrn[2];
    auto load_qsz = [&](int kt) {
#pragma unroll
        for (int t = 0; t < 2; ++t) {
            qn[t]  = *(const int2*)(qrow[t] + kt * (KTILE / 2));
            scn[t] = *(const uint2*)(scp[t] + (size_t)kt * NDIM);
            zrn[t] = *(const uint2*)(zrp[t] + (size_t)kt * NDIM);
        }
    };

    auto dequant = [&](int buf, const int2* q, const uint2* sc, const uint2* zr) {
#pragma unroll
        for (int t = 0; t < 2; ++t) {
            unsigned short* wr = &Ws[buf][o4r[t] * 4][pr[t] * 4];
            unsigned short sb[4] = { (unsigned short)(sc[t].x & 0xFFFF), (unsigned short)(sc[t].x >> 16),
                                     (unsigned short)(sc[t].y & 0xFFFF), (unsigned short)(sc[t].y >> 16) };
            unsigned short zb[4] = { (unsigned short)(zr[t].x & 0xFFFF), (unsigned short)(zr[t].x >> 16),
                                     (unsigned short)(zr[t].y & 0xFFFF), (unsigned short)(zr[t].y >> 16) };
            int qx = q[t].x, qy = q[t].y;
#pragma unroll
            for (int a = 0; a < 4; ++a) {          // row within the 4-row pack
                float s = bf2f(sb[a]);
                float z = bf2f(zb[a]);
                // nibble (a*2+b)*4 -> W[4*o4+a][2*i2+b]
                float w0 = (float)((qx >> (8 * a))     & 0xF);
                float w1 = (float)((qx >> (8 * a + 4)) & 0xF);
                float w2 = (float)((qy >> (8 * a))     & 0xF);
                float w3 = (float)((qy >> (8 * a + 4)) & 0xF);
                uint2 v;
                v.x = (uint32_t)f2bf((w0 - z) * s) | ((uint32_t)f2bf((w1 - z) * s) << 16);
                v.y = (uint32_t)f2bf((w2 - z) * s) | ((uint32_t)f2bf((w3 - z) * s) << 16);
                *(uint2*)(wr + a * LDST) = v;      // ds_store_b64
            }
        }
    };

    v8f acc[4][2] = {};

    auto compute = [&](int buf) {
#pragma unroll
        for (int ks = 0; ks < 2; ++ks) {           // two 16x16x32 K-steps per K-tile
            const int kb = ks * 32;
            v16bf av[4], bv[2];
#pragma unroll
            for (int fm = 0; fm < 4; ++fm) {
                const unsigned short* p = &As[buf][wm * 64 + fm * 16 + r16][kb + hh * 8];
                v8bf c0 = *(const v8bf*)p;         // K = kb + {0..7}/{8..15}
                v8bf c1 = *(const v8bf*)(p + 16);  // K = kb + 16 + ...
                av[fm] = __builtin_shufflevector(c0, c1, 0,1,2,3,4,5,6,7,8,9,10,11,12,13,14,15);
            }
#pragma unroll
            for (int fn = 0; fn < 2; ++fn) {
                const unsigned short* p = &Ws[buf][wn * 32 + fn * 16 + r16][kb + hh * 8];
                v8bf d0 = *(const v8bf*)p;         // K = kb + {0..7}/{8..15}, then kb + 16 + ...
                v8bf d1 = *(const v8bf*)(p + 16);
                bv[fn] = __builtin_shufflevector(d0, d1, 0,1,2,3,4,5,6,7,8,9,10,11,12,13,14,15);
            }
#pragma unroll
            for (int fm = 0; fm < 4; ++fm)
#pragma unroll
                for (int fn = 0; fn < 2; ++fn)
                    acc[fm][fn] = __builtin_amdgcn_wmma_f32_16x16x32_bf16(
                        false, av[fm], false, bv[fn], (short)0, acc[fm][fn], false, false);
        }
    };

    // ---- software pipeline ----
    issue_a(0, 0);
    load_qsz(0);
    int2  qc[2];
    uint2 scc[2], zrc[2];

    for (int kt = 0; kt < NK; ++kt) {
        const int buf = kt & 1;
#pragma unroll
        for (int t = 0; t < 2; ++t) { qc[t] = qn[t]; scc[t] = scn[t]; zrc[t] = zrn[t]; }

        dequant(buf, qc, scc, zrc);
#ifdef USE_ASYNC
        __builtin_amdgcn_s_wait_asynccnt(0);
#endif
        __syncthreads();

        if (kt + 1 < NK) {
            issue_a(kt + 1, buf ^ 1);
            load_qsz(kt + 1);
        }
        compute(buf);
    }

    // ---- epilogue: bias + bf16 store ----
#pragma unroll
    for (int fm = 0; fm < 4; ++fm) {
        const int row = m0 + wm * 64 + fm * 16 + hh * 8;
#pragma unroll
        for (int fn = 0; fn < 2; ++fn) {
            const int col = n0 + wn * 32 + fn * 16 + r16;
            const float b = bf2f(bias[col]);
            unsigned short* op = out + (size_t)row * NDIM + col;
#pragma unroll
            for (int e = 0; e < 8; ++e)
                op[(size_t)e * NDIM] = f2bf(acc[fm][fn][e] + b);
        }
    }
}

extern "C" void kernel_launch(void* const* d_in, const int* in_sizes, int n_in,
                              void* d_out, int out_size, void* d_ws, size_t ws_size,
                              hipStream_t stream) {
    const unsigned short* x  = (const unsigned short*)d_in[0];
    const int*            q  = (const int*)d_in[1];
    const unsigned short* sc = (const unsigned short*)d_in[2];
    const unsigned short* zr = (const unsigned short*)d_in[3];
    const unsigned short* bs = (const unsigned short*)d_in[4];
    unsigned short* out = (unsigned short*)d_out;

    dim3 grid(NDIM / NTILE, MDIM / MTILE);   // (96, 32)
    awq_w4a16_wmma<<<grid, 256, 0, stream>>>(x, q, sc, zr, bs, out);
}
